// SecondOrderDeformableAlignment_10866267259444
// MI455X (gfx1250) — matrix operand order repaired and checked by reading, not compiled, not run on hardware
//
#include <hip/hip_runtime.h>
#include <hip/hip_bf16.h>
#include <math.h>

// ---------------------------------------------------------------------------
// SecondOrderDeformableAlignment on gfx1250 (MI455X).
// All GEMMs (4 convs + deformable conv) run on v_wmma_f32_16x16x32_f16 with
// register tiling (MTILE x NTILE 16x16 tiles per wave). Deformable sampling
// uses branch-free hardware transcendentals (v_exp_f32 / v_rcp_f32).
// ---------------------------------------------------------------------------

typedef __attribute__((ext_vector_type(16))) _Float16 v16h;
typedef __attribute__((ext_vector_type(8)))  _Float16 half8;
typedef __attribute__((ext_vector_type(2)))  _Float16 half2;
typedef __attribute__((ext_vector_type(8)))  float    v8f;

#define HH   128
#define WW   128
#define HW   16384           // 128*128
#define BB   2
#define NPIX 32768           // B*H*W

// ------------------------- layout prep kernels -----------------------------

// NCHW f32 -> NHWC f16
__global__ void nchw_to_nhwc_h(const float* __restrict__ in,
                               _Float16* __restrict__ out, int C, int n) {
  int i = blockIdx.x * blockDim.x + threadIdx.x;
  if (i >= n) return;
  int x = i % WW;  int t = i / WW;
  int y = t % HH;  t /= HH;
  int c = t % C;   int b = t / C;
  out[((b * HW + y * WW + x) * C) + c] = (_Float16)in[i];
}

// w[o][c][3][3] f32 -> wB[o][K] f16 with K = j*Cin + c  (j = ky*3+kx)
__global__ void prep_w_conv(const float* __restrict__ w,
                            _Float16* __restrict__ out, int Cin, int n) {
  int i = blockIdx.x * blockDim.x + threadIdx.x;
  if (i >= n) return;
  int j = i % 9;  int t = i / 9;
  int c = t % Cin; int o = t / Cin;
  out[o * (9 * Cin) + j * Cin + c] = (_Float16)w[i];
}

// dw[o][128][3][3] f32 -> wBd[o][1152] f16 with K = (g*9+j)*8 + cc
__global__ void prep_w_def(const float* __restrict__ w,
                           _Float16* __restrict__ out, int n) {
  int i = blockIdx.x * blockDim.x + threadIdx.x;
  if (i >= n) return;
  int j = i % 9;  int t = i / 9;
  int cg = t % 128; int o = t / 128;
  int g = cg >> 3, cc = cg & 7;
  out[o * 1152 + (g * 9 + j) * 8 + cc] = (_Float16)w[i];
}

// --------------------------- conv3x3 (implicit GEMM) -----------------------
// One wave per block computes (16*MTILE) pixels x (16*NTILE) out channels.
// feat: NHWC f16 [B,H,W,Cin], wB: [Cout][9*Cin] f16 (K = j*Cin+c).
// Requires Cin % 32 == 0.
template <int MTILE, int NTILE>
__global__ void __launch_bounds__(32)
conv3x3_wmma(const _Float16* __restrict__ feat,
             const _Float16* __restrict__ wB,
             const float* __restrict__ bias,
             _Float16* __restrict__ out,
             int Cin, int Cout, int do_lrelu) {
  const int lane = threadIdx.x;
  const int m    = lane & 15;       // A row / B column within a 16x16 tile
  const int hi   = lane >> 4;       // half-wave selector
  const int pbase = blockIdx.x * (16 * MTILE);
  const int n0    = blockIdx.y * (16 * NTILE);
  const int Ktot  = 9 * Cin;

  // per-M-tile pixel coordinates for this lane's A row
  int yb[MTILE], xb[MTILE], bb[MTILE];
#pragma unroll
  for (int mt = 0; mt < MTILE; ++mt) {
    const int p = pbase + mt * 16 + m;
    bb[mt] = p >> 14;
    const int rr = p & (HW - 1);
    yb[mt] = rr >> 7;
    xb[mt] = rr & (WW - 1);
  }

  v8f acc[MTILE][NTILE];
#pragma unroll
  for (int mt = 0; mt < MTILE; ++mt)
#pragma unroll
    for (int t = 0; t < NTILE; ++t) acc[mt][t] = (v8f){};

  for (int j = 0; j < 9; ++j) {
    const int dy = j / 3 - 1, dx = j % 3 - 1;
    bool valid[MTILE];
    const _Float16* src[MTILE];
#pragma unroll
    for (int mt = 0; mt < MTILE; ++mt) {
      const int yy = yb[mt] + dy, xx = xb[mt] + dx;
      valid[mt] = (yy >= 0) & (yy < HH) & (xx >= 0) & (xx < WW);
      src[mt] = valid[mt] ? feat + ((bb[mt] * HW + yy * WW + xx) * Cin) : feat;
    }

    for (int c0 = 0; c0 < Cin; c0 += 32) {
      // ---- A fragments: row m; K-in-step 16 contiguous at hi*16
      v16h a[MTILE];
#pragma unroll
      for (int mt = 0; mt < MTILE; ++mt) {
        a[mt] = (v16h){};
        if (valid[mt]) {
          const _Float16* ap = src[mt] + c0 + hi * 16;
          half8 lo = *(const half8*)(ap);
          half8 hb = *(const half8*)(ap + 8);
          a[mt] = __builtin_shufflevector(lo, hb, 0, 1, 2, 3, 4, 5, 6, 7,
                                          8, 9, 10, 11, 12, 13, 14, 15);
        }
      }
      // ---- B fragments: column n0+t*16+m, 16 contiguous K at hi*16
      v16h bf[NTILE];
#pragma unroll
      for (int t = 0; t < NTILE; ++t) {
        const _Float16* wp =
            wB + (n0 + t * 16 + m) * Ktot + (j * Cin + c0) + hi * 16;
        half8 b0 = *(const half8*)(wp);
        half8 b1 = *(const half8*)(wp + 8);
        bf[t] = __builtin_shufflevector(b0, b1, 0, 1, 2, 3, 4, 5, 6, 7,
                                        8, 9, 10, 11, 12, 13, 14, 15);
      }
      // ---- independent WMMA chain (MTILE*NTILE per K-step)
#pragma unroll
      for (int mt = 0; mt < MTILE; ++mt)
#pragma unroll
        for (int t = 0; t < NTILE; ++t)
          acc[mt][t] = __builtin_amdgcn_wmma_f32_16x16x32_f16(
              false, a[mt], false, bf[t], (short)0, acc[mt][t], false, false);
    }
  }

  // epilogue: bias (+ leaky relu), NHWC f16 store
#pragma unroll
  for (int mt = 0; mt < MTILE; ++mt) {
#pragma unroll
    for (int t = 0; t < NTILE; ++t) {
      const int n = n0 + t * 16 + m;
      const float bn = bias[n];
#pragma unroll
      for (int r = 0; r < 8; ++r) {
        float v = acc[mt][t][r] + bn;
        if (do_lrelu) v = (v >= 0.f) ? v : 0.1f * v;
        const int po = pbase + mt * 16 + r + hi * 8;   // C/D row M = r + hi*8
        out[po * Cout + n] = (_Float16)v;
      }
    }
  }
}

// ------------------- modulated deformable conv (DCNv2) ---------------------
// Branch-free transcendentals on the hardware TRANS pipe.
__device__ inline float fast_tanh(float x) {
  const float t = __expf(-2.0f * fabsf(x));          // v_exp_f32 path
  const float r = (1.0f - t) * __builtin_amdgcn_rcpf(1.0f + t);
  return copysignf(r, x);
}
__device__ inline float fast_sigmoid(float x) {
  return __builtin_amdgcn_rcpf(1.0f + __expf(-x));
}

// K = (g*9+j)*8 + cc, Ktot = 1152. A-fragment halves are built by bilinear
// sampling of 8 contiguous group channels per (g,j) slot, modulated by mask.
__device__ inline void sample_slot(const _Float16* __restrict__ xh,
                                   const _Float16* __restrict__ c4p,
                                   int b, int y, int x, int slot,
                                   float* __restrict__ v) {
  const int g = slot / 9;
  const int j = slot - g * 9;
  const half2 dxy = *(const half2*)(c4p + slot * 2);   // one b32 load
  const float dyo = 5.0f * fast_tanh((float)dxy[0]);
  const float dxo = 5.0f * fast_tanh((float)dxy[1]);
  const float mk  = fast_sigmoid((float)c4p[288 + slot]);

  const float ys = (float)(y + j / 3 - 1) + dyo;
  const float xs = (float)(x + j % 3 - 1) + dxo;
  const float y0f = floorf(ys), x0f = floorf(xs);
  const float wy = ys - y0f, wx = xs - x0f;
  const int y0 = (int)y0f, x0 = (int)x0f;

  float vv[8] = {0.f, 0.f, 0.f, 0.f, 0.f, 0.f, 0.f, 0.f};
  const int cbase = g * 8;

#pragma unroll
  for (int cy = 0; cy < 2; ++cy) {
#pragma unroll
    for (int cx = 0; cx < 2; ++cx) {
      const int yi = y0 + cy, xi = x0 + cx;
      const float wgt = (cy ? wy : 1.f - wy) * (cx ? wx : 1.f - wx);
      if (yi >= 0 && yi < HH && xi >= 0 && xi < WW) {
        const half8 px =
            *(const half8*)(xh + ((b * HW + yi * WW + xi) * 128 + cbase));
#pragma unroll
        for (int c = 0; c < 8; ++c) vv[c] += wgt * (float)px[c];
      }
    }
  }
#pragma unroll
  for (int c = 0; c < 8; ++c) v[c] = vv[c] * mk;
}

__global__ void __launch_bounds__(32)
deform_wmma(const _Float16* __restrict__ xh,    // NHWC f16 [B,H,W,128]
            const _Float16* __restrict__ c4,    // NHWC f16 [B,H,W,432]
            const _Float16* __restrict__ wBd,   // [64][1152] f16
            const float* __restrict__ db,
            float* __restrict__ out) {          // NCHW f32 [B,64,H,W]
  constexpr int NTILE = 4;                      // 64 out channels
  const int lane = threadIdx.x;
  const int m    = lane & 15;
  const int hi   = lane >> 4;
  const int pbase = blockIdx.x * 16;

  const int p = pbase + m;
  const int b = p >> 14;
  const int rr = p & (HW - 1);
  const int y = rr >> 7;
  const int x = rr & (WW - 1);
  const _Float16* c4p = c4 + p * 432;

  v8f acc[NTILE];
#pragma unroll
  for (int t = 0; t < NTILE; ++t) acc[t] = (v8f){};

  for (int s = 0; s < 36; ++s) {            // 36 * 32 = 1152 = Ktot
    // A fragment: this lane covers slots s*4+hi (halves 0..7)
    //             and s*4+2+hi (halves 8..15)
    float va[8], vb[8];
    sample_slot(xh, c4p, b, y, x, s * 4 + hi, va);
    sample_slot(xh, c4p, b, y, x, s * 4 + 2 + hi, vb);
    v16h a;
#pragma unroll
    for (int c = 0; c < 8; ++c) {
      a[c]     = (_Float16)va[c];
      a[c + 8] = (_Float16)vb[c];
    }
    // B fragments (4 N tiles), then 4 independent WMMAs sharing A
#pragma unroll
    for (int t = 0; t < NTILE; ++t) {
      const _Float16* wp = wBd + (t * 16 + m) * 1152 + s * 32 + hi * 8;
      half8 b0 = *(const half8*)(wp);
      half8 b1 = *(const half8*)(wp + 16);
      v16h bf = __builtin_shufflevector(b0, b1, 0, 1, 2, 3, 4, 5, 6, 7,
                                        8, 9, 10, 11, 12, 13, 14, 15);
      acc[t] = __builtin_amdgcn_wmma_f32_16x16x32_f16(
          false, a, false, bf, (short)0, acc[t], false, false);
    }
  }

#pragma unroll
  for (int t = 0; t < NTILE; ++t) {
    const int n = t * 16 + m;
    const float bn = db[n];
#pragma unroll
    for (int r = 0; r < 8; ++r) {
      const int po = pbase + r + hi * 8;
      const int bo = po >> 14;
      const int rro = po & (HW - 1);
      out[(bo * 64 + n) * HW + rro] = acc[t][r] + bn;
    }
  }
}

// ------------------------------- launcher ----------------------------------

extern "C" void kernel_launch(void* const* d_in, const int* in_sizes, int n_in,
                              void* d_out, int out_size, void* d_ws, size_t ws_size,
                              hipStream_t stream) {
  const float* x   = (const float*)d_in[0];   // (2,128,128,128)
  const float* ef  = (const float*)d_in[1];   // (2,192,128,128)
  const float* w1  = (const float*)d_in[2];
  const float* b1  = (const float*)d_in[3];
  const float* w2  = (const float*)d_in[4];
  const float* b2  = (const float*)d_in[5];
  const float* w3  = (const float*)d_in[6];
  const float* b3  = (const float*)d_in[7];
  const float* w4  = (const float*)d_in[8];   // (432,64,3,3)
  const float* b4  = (const float*)d_in[9];
  const float* dw  = (const float*)d_in[10];  // (64,128,3,3)
  const float* db  = (const float*)d_in[11];
  float* out = (float*)d_out;

  // workspace carve-out (halves), 256B-aligned regions
  char* ws = (char*)d_ws;
  size_t off = 0;
  auto take = [&](size_t halves) {
    _Float16* p = (_Float16*)(ws + off);
    off += (halves * 2 + 255) & ~(size_t)255;
    return p;
  };
  _Float16* xh  = take((size_t)NPIX * 128);   // x NHWC f16
  _Float16* efh = take((size_t)NPIX * 192);   // extra_feat NHWC f16
  _Float16* h0  = take((size_t)NPIX * 64);
  _Float16* h1  = take((size_t)NPIX * 64);
  _Float16* c4  = take((size_t)NPIX * 432);   // conv4 raw output NHWC f16
  _Float16* wB1 = take((size_t)64 * 1728);
  _Float16* wB2 = take((size_t)64 * 576);
  _Float16* wB3 = take((size_t)64 * 576);
  _Float16* wB4 = take((size_t)432 * 576);
  _Float16* wBd = take((size_t)64 * 1152);

  const int TB = 256;
  auto grid1 = [&](int n) { return dim3((n + TB - 1) / TB); };

  // layout conversions
  nchw_to_nhwc_h<<<grid1(NPIX * 128), TB, 0, stream>>>(x, xh, 128, NPIX * 128);
  nchw_to_nhwc_h<<<grid1(NPIX * 192), TB, 0, stream>>>(ef, efh, 192, NPIX * 192);
  prep_w_conv<<<grid1(64 * 1728), TB, 0, stream>>>(w1, wB1, 192, 64 * 1728);
  prep_w_conv<<<grid1(64 * 576),  TB, 0, stream>>>(w2, wB2, 64, 64 * 576);
  prep_w_conv<<<grid1(64 * 576),  TB, 0, stream>>>(w3, wB3, 64, 64 * 576);
  prep_w_conv<<<grid1(432 * 576), TB, 0, stream>>>(w4, wB4, 64, 432 * 576);
  prep_w_def<<<grid1(64 * 1152),  TB, 0, stream>>>(dw, wBd, 64 * 1152);

  // conv stack (implicit GEMM, WMMA, register-tiled)
  // conv1-3: 32 pixels x 64 outch per wave; conv4: 32 pixels x 48 outch.
  conv3x3_wmma<2, 4><<<dim3(NPIX / 32, 1), 32, 0, stream>>>(
      efh, wB1, b1, h0, 192, 64, 1);
  conv3x3_wmma<2, 4><<<dim3(NPIX / 32, 1), 32, 0, stream>>>(
      h0, wB2, b2, h1, 64, 64, 1);
  conv3x3_wmma<2, 4><<<dim3(NPIX / 32, 1), 32, 0, stream>>>(
      h1, wB3, b3, h0, 64, 64, 1);
  conv3x3_wmma<2, 3><<<dim3(NPIX / 32, 9), 32, 0, stream>>>(
      h0, wB4, b4, c4, 64, 432, 0);

  // modulated deformable conv (offsets/mask decoded inline from c4)
  deform_wmma<<<dim3(NPIX / 16, 1), 32, 0, stream>>>(xh, c4, wBd, db, out);
}
